// Seq2Seq_49091476193938
// MI455X (gfx1250) — matrix operand order repaired and checked by reading, not compiled, not run on hardware
//
#include <hip/hip_runtime.h>
#include <cstddef>

#define BATCH   64
#define EMB     256
#define HID     1024
#define GATES   4096   // 4*HID
#define VOCAB   32000
#define SRC_LEN 64
#define TRG_LEN 64

typedef __attribute__((ext_vector_type(16))) __bf16 bf16x16;
typedef __attribute__((ext_vector_type(8)))  __bf16 bf16x8;
typedef __attribute__((ext_vector_type(8)))  float  f32x8;

// float -> bf16 round-to-nearest-even via bit manipulation.
__device__ __forceinline__ __bf16 f2bf(float f) {
    unsigned int u = __builtin_bit_cast(unsigned int, f);
    u += 0x7FFFu + ((u >> 16) & 1u);
    unsigned short s = (unsigned short)(u >> 16);
    return __builtin_bit_cast(__bf16, s);
}

__device__ __forceinline__ float sigm(float x) {
    return 1.0f / (1.0f + __expf(-x));
}

// A-fragment loader for V_WMMA_*_16X16X32 (16-bit A, 16x32):
// lane holds row M = lane&15; K chunk {2*aklo..+15} (contiguous),
// aklo = (lane>>4)*8. Two contiguous 16-byte loads.
__device__ __forceinline__ bf16x16 load_a_frag(const __bf16* __restrict__ p, int aklo) {
    bf16x8 lo = *(const bf16x8*)(p + 2 * aklo);
    bf16x8 hi = *(const bf16x8*)(p + 2 * aklo + 8);
    bf16x16 a;
#pragma unroll
    for (int e = 0; e < 8; ++e) { a[e] = lo[e]; a[e + 8] = hi[e]; }
    return a;
}

__device__ __forceinline__ f32x8 wmma_bf16(bf16x16 a, bf16x16 b, f32x8 c) {
    return __builtin_amdgcn_wmma_f32_16x16x32_bf16(false, a, false, b,
                                                   (short)0, c, false, false);
}

// ---------------------------------------------------------------------------
// One-time f32 -> bf16 conversion (weights AND embedding tables). fc_W bf16 =
// 65.5 MB -> resident in the 192 MB L2 across all 63 decoder steps.
// ---------------------------------------------------------------------------
__global__ __launch_bounds__(256)
void cvt_bf16_kernel(const float* __restrict__ src, __bf16* __restrict__ dst, long n) {
    long i = (long)blockIdx.x * 256 + threadIdx.x;
    long stride = (long)gridDim.x * 256;
    for (; i < n; i += stride) dst[i] = f2bf(src[i]);
}

__global__ __launch_bounds__(256)
void bias_combine_kernel(const float* __restrict__ b_ih, const float* __restrict__ b_hh,
                         float* __restrict__ out, int n) {
    int i = blockIdx.x * 256 + threadIdx.x;
    if (i < n) out[i] = b_ih[i] + b_hh[i];
}

// ---------------------------------------------------------------------------
// Fused LSTM step, register-resident gates:
//   gates[64,4096] = emb_bf[tok][64,256] @ W_ih^T + h[64,1024] @ W_hh^T + b
// Block = 128 threads = 4 waves; wave wv owns M-tile wv and ALL FOUR gate
// tiles (i,f,g,o) for hidden slice blockIdx.x. Since element (m,n) occupies
// the same lane/slot in every accumulator, the LSTM nonlinearity runs fully
// in registers: no LDS, no barrier on the serial critical path. 4 independent
// WMMA chains per wave give the scheduler loads to overlap with matrix ops.
// ---------------------------------------------------------------------------
__global__ __launch_bounds__(128)
void lstm_step_kernel(const int*    __restrict__ tok,     // [64]
                      const __bf16* __restrict__ emb_bf,  // [V, 256]
                      const __bf16* __restrict__ Wih,     // [4096, 256]
                      const __bf16* __restrict__ Whh,     // [4096, 1024]
                      const float*  __restrict__ bias,    // [4096] (b_ih+b_hh)
                      const __bf16* __restrict__ h_in,    // [64, 1024]
                      __bf16*       __restrict__ h_out,   // [64, 1024]
                      float*        __restrict__ c)       // [64, 1024] in/out
{
    const int tid  = threadIdx.x;
    const int lane = tid & 31;
    const int wv   = tid >> 5;        // M tile 0..3
    const int nblk = blockIdx.x;      // hidden slice [16*nblk, 16*nblk+16)
    const int l15  = lane & 15;
    const int lhi  = lane >> 4;
    const int mrow = wv * 16 + l15;   // A row this lane feeds
    const int aklo = lhi * 8;         // A layout K sub-offset
    const int bklo = lhi * 16;        // B layout K sub-offset
    const int ncol = nblk * 16 + l15; // B/D column within each gate

    const int token = tok[mrow];
    const __bf16* xrow = emb_bf + (size_t)token * EMB;
    const __bf16* hrow = h_in + (size_t)mrow * HID;

    f32x8 acc0, acc1, acc2, acc3;     // i, f, g, o accumulators
    {
        const float b0 = bias[0 * HID + ncol];
        const float b1 = bias[1 * HID + ncol];
        const float b2 = bias[2 * HID + ncol];
        const float b3 = bias[3 * HID + ncol];
#pragma unroll
        for (int r = 0; r < 8; ++r) {
            acc0[r] = b0; acc1[r] = b1; acc2[r] = b2; acc3[r] = b3;
        }
    }

    const __bf16* WihR0 = Wih + (size_t)(0 * HID + ncol) * EMB;
    const __bf16* WihR1 = Wih + (size_t)(1 * HID + ncol) * EMB;
    const __bf16* WihR2 = Wih + (size_t)(2 * HID + ncol) * EMB;
    const __bf16* WihR3 = Wih + (size_t)(3 * HID + ncol) * EMB;
    const __bf16* WhhR0 = Whh + (size_t)(0 * HID + ncol) * HID;
    const __bf16* WhhR1 = Whh + (size_t)(1 * HID + ncol) * HID;
    const __bf16* WhhR2 = Whh + (size_t)(2 * HID + ncol) * HID;
    const __bf16* WhhR3 = Whh + (size_t)(3 * HID + ncol) * HID;

    // --- x @ W_ih^T : K = 256, 8 k-steps x 4 gate WMMAs ---
#pragma unroll
    for (int k32 = 0; k32 < EMB / 32; ++k32) {
        const int kb = k32 * 32;
        bf16x16 a  = load_a_frag(xrow + kb, aklo);
        bf16x16 b0 = *(const bf16x16*)(WihR0 + kb + bklo);
        bf16x16 b1 = *(const bf16x16*)(WihR1 + kb + bklo);
        bf16x16 b2 = *(const bf16x16*)(WihR2 + kb + bklo);
        bf16x16 b3 = *(const bf16x16*)(WihR3 + kb + bklo);
        acc0 = wmma_bf16(a, b0, acc0);
        acc1 = wmma_bf16(a, b1, acc1);
        acc2 = wmma_bf16(a, b2, acc2);
        acc3 = wmma_bf16(a, b3, acc3);
    }

    // --- h @ W_hh^T : K = 1024, 32 k-steps x 4 gate WMMAs ---
#pragma unroll 4
    for (int k32 = 0; k32 < HID / 32; ++k32) {
        const int kb = k32 * 32;
        bf16x16 a  = load_a_frag(hrow + kb, aklo);
        bf16x16 b0 = *(const bf16x16*)(WhhR0 + kb + bklo);
        bf16x16 b1 = *(const bf16x16*)(WhhR1 + kb + bklo);
        bf16x16 b2 = *(const bf16x16*)(WhhR2 + kb + bklo);
        bf16x16 b3 = *(const bf16x16*)(WhhR3 + kb + bklo);
        acc0 = wmma_bf16(a, b0, acc0);
        acc1 = wmma_bf16(a, b1, acc1);
        acc2 = wmma_bf16(a, b2, acc2);
        acc3 = wmma_bf16(a, b3, acc3);
    }

    // --- LSTM nonlinearity, fully in registers (i,f,g,o share lane/slot) ---
    const int mbase = wv * 16 + lhi * 8;   // D-layout: VGPR r -> row mbase+r
#pragma unroll
    for (int r = 0; r < 8; ++r) {
        const int m = mbase + r;
        const float co = c[(size_t)m * HID + ncol];
        const float cn = sigm(acc1[r]) * co + sigm(acc0[r]) * tanhf(acc2[r]);
        const float hn = sigm(acc3[r]) * tanhf(cn);
        c[(size_t)m * HID + ncol]     = cn;
        h_out[(size_t)m * HID + ncol] = f2bf(hn);
    }
}

// ---------------------------------------------------------------------------
// logits[64,32000] = h @ fc_W^T + fc_b   (M=64, N=32000, K=1024)
// Each wave computes TWO 16x16 N-tiles sharing one A-fragment (2 independent
// WMMA chains, halved A traffic). 4000 wave-pairs / 8 waves -> 500 blocks.
// fc_W bf16 stays L2-resident across the 63 decoder steps; global_prefetch
// pulls each weight row toward the WGP ahead of use.
// ---------------------------------------------------------------------------
__global__ __launch_bounds__(256)
void logits_kernel(const __bf16* __restrict__ h_bf,   // [64, 1024]
                   const __bf16* __restrict__ fcW,    // [32000, 1024]
                   const float*  __restrict__ fcb,    // [32000]
                   float*        __restrict__ out)    // [64, 32000]
{
    const int tid  = threadIdx.x;
    const int lane = tid & 31;
    const int wv   = tid >> 5;
    const int wg   = blockIdx.x * 8 + wv;   // 0..3999
    const int mt   = wg & 3;
    const int ntp  = wg >> 2;               // 0..999 -> N tiles 2*ntp, 2*ntp+1
    const int l15  = lane & 15;
    const int lhi  = lane >> 4;
    const int mrow = mt * 16 + l15;
    const int nc0  = (2 * ntp) * 16 + l15;
    const int nc1  = nc0 + 16;
    const __bf16* hrow  = h_bf + (size_t)mrow * HID;
    const __bf16* wrow0 = fcW + (size_t)nc0 * HID;
    const __bf16* wrow1 = fcW + (size_t)nc1 * HID;

    f32x8 acc0, acc1;
    {
        const float bb0 = fcb[nc0];
        const float bb1 = fcb[nc1];
#pragma unroll
        for (int r = 0; r < 8; ++r) { acc0[r] = bb0; acc1[r] = bb1; }
    }

    // Pull this wave's weight rows toward the WGP (global_prefetch_b8).
    __builtin_prefetch((const void*)(wrow0 + 512), 0, 1);
    __builtin_prefetch((const void*)(wrow1 + 512), 0, 1);

#pragma unroll 4
    for (int k32 = 0; k32 < HID / 32; ++k32) {
        const int kb = k32 * 32;
        bf16x16 a  = load_a_frag(hrow + kb, lhi * 8);
        bf16x16 b0 = *(const bf16x16*)(wrow0 + kb + lhi * 16);
        bf16x16 b1 = *(const bf16x16*)(wrow1 + kb + lhi * 16);
        acc0 = wmma_bf16(a, b0, acc0);
        acc1 = wmma_bf16(a, b1, acc1);
    }

    const int mb = mt * 16 + lhi * 8;
#pragma unroll
    for (int r = 0; r < 8; ++r) {
        out[(size_t)(mb + r) * VOCAB + nc0] = acc0[r];
        out[(size_t)(mb + r) * VOCAB + nc1] = acc1[r];
    }
}

// ---------------------------------------------------------------------------
// Per-row argmax over 32000 logits -> next token (first-max tie-break).
// ---------------------------------------------------------------------------
__global__ __launch_bounds__(256)
void argmax_kernel(const float* __restrict__ logits, int* __restrict__ tok_out) {
    const int m = blockIdx.x;
    const float* row = logits + (size_t)m * VOCAB;
    float best = -__builtin_inff();
    int   bidx = 0;
    for (int i = threadIdx.x; i < VOCAB; i += 256) {
        float v = row[i];
        if (v > best) { best = v; bidx = i; }   // ascending scan -> first max
    }
    __shared__ float sv[256];
    __shared__ int   si[256];
    sv[threadIdx.x] = best; si[threadIdx.x] = bidx;
    __syncthreads();
    for (int s = 128; s > 0; s >>= 1) {
        if (threadIdx.x < s) {
            float ov = sv[threadIdx.x + s];
            int   oi = si[threadIdx.x + s];
            if (ov > sv[threadIdx.x] ||
                (ov == sv[threadIdx.x] && oi < si[threadIdx.x])) {
                sv[threadIdx.x] = ov; si[threadIdx.x] = oi;
            }
        }
        __syncthreads();
    }
    if (threadIdx.x == 0) tok_out[m] = si[0];
}

// ---------------------------------------------------------------------------
extern "C" void kernel_launch(void* const* d_in, const int* in_sizes, int n_in,
                              void* d_out, int out_size, void* d_ws, size_t ws_size,
                              hipStream_t stream) {
    (void)in_sizes; (void)n_in; (void)out_size; (void)ws_size;
    const int*   src     = (const int*)  d_in[0];
    const int*   trg     = (const int*)  d_in[1];
    const float* enc_emb = (const float*)d_in[2];
    const float* encWih  = (const float*)d_in[3];
    const float* encWhh  = (const float*)d_in[4];
    const float* enc_bih = (const float*)d_in[5];
    const float* enc_bhh = (const float*)d_in[6];
    const float* dec_emb = (const float*)d_in[7];
    const float* decWih  = (const float*)d_in[8];
    const float* decWhh  = (const float*)d_in[9];
    const float* dec_bih = (const float*)d_in[10];
    const float* dec_bhh = (const float*)d_in[11];
    const float* fcW     = (const float*)d_in[12];
    const float* fcb     = (const float*)d_in[13];
    float* out = (float*)d_out;

    // Workspace carve-up (all offsets 256B aligned). Total ~120 MB.
    size_t off = 0;
    auto carve = [&](size_t bytes) -> void* {
        void* p = (char*)d_ws + off;
        off += (bytes + 255) & ~(size_t)255;
        return p;
    };
    __bf16* encWih_bf  = (__bf16*)carve((size_t)GATES * EMB * 2);
    __bf16* encWhh_bf  = (__bf16*)carve((size_t)GATES * HID * 2);
    __bf16* decWih_bf  = (__bf16*)carve((size_t)GATES * EMB * 2);
    __bf16* decWhh_bf  = (__bf16*)carve((size_t)GATES * HID * 2);
    __bf16* fcW_bf     = (__bf16*)carve((size_t)VOCAB * HID * 2);
    __bf16* encEmb_bf  = (__bf16*)carve((size_t)VOCAB * EMB * 2);
    __bf16* decEmb_bf  = (__bf16*)carve((size_t)VOCAB * EMB * 2);
    float*  enc_bias   = (float*) carve((size_t)GATES * 4);
    float*  dec_bias   = (float*) carve((size_t)GATES * 4);
    __bf16* hA         = (__bf16*)carve((size_t)BATCH * HID * 2);
    __bf16* hB         = (__bf16*)carve((size_t)BATCH * HID * 2);
    float*  cbuf       = (float*) carve((size_t)BATCH * HID * 4);
    int*    ntok       = (int*)   carve((size_t)BATCH * 4);

    // One-time precompute: weights + embeddings -> bf16, biases combined.
    cvt_bf16_kernel<<<2048, 256, 0, stream>>>(encWih, encWih_bf, (long)GATES * EMB);
    cvt_bf16_kernel<<<2048, 256, 0, stream>>>(encWhh, encWhh_bf, (long)GATES * HID);
    cvt_bf16_kernel<<<2048, 256, 0, stream>>>(decWih, decWih_bf, (long)GATES * EMB);
    cvt_bf16_kernel<<<2048, 256, 0, stream>>>(decWhh, decWhh_bf, (long)GATES * HID);
    cvt_bf16_kernel<<<4096, 256, 0, stream>>>(fcW,    fcW_bf,    (long)VOCAB * HID);
    cvt_bf16_kernel<<<2048, 256, 0, stream>>>(enc_emb, encEmb_bf, (long)VOCAB * EMB);
    cvt_bf16_kernel<<<2048, 256, 0, stream>>>(dec_emb, decEmb_bf, (long)VOCAB * EMB);
    bias_combine_kernel<<<16, 256, 0, stream>>>(enc_bih, enc_bhh, enc_bias, GATES);
    bias_combine_kernel<<<16, 256, 0, stream>>>(dec_bih, dec_bhh, dec_bias, GATES);

    // h0 = 0, c0 = 0, outputs[0] = 0.
    hipMemsetAsync(hA,   0, (size_t)BATCH * HID * 2, stream);
    hipMemsetAsync(cbuf, 0, (size_t)BATCH * HID * 4, stream);
    hipMemsetAsync(out,  0, (size_t)BATCH * VOCAB * 4, stream);

    __bf16* hp = hA;  // h_in
    __bf16* hn = hB;  // h_out

    // Encoder: 64 sequential steps.
    for (int t = 0; t < SRC_LEN; ++t) {
        lstm_step_kernel<<<HID / 16, 128, 0, stream>>>(
            src + (size_t)t * BATCH, encEmb_bf, encWih_bf, encWhh_bf,
            enc_bias, hp, hn, cbuf);
        __bf16* tmp = hp; hp = hn; hn = tmp;
    }

    // Decoder: greedy autoregressive, 63 steps (teacher forcing off).
    const int* tokp = trg;  // trg[0]
    for (int t = 1; t < TRG_LEN; ++t) {
        lstm_step_kernel<<<HID / 16, 128, 0, stream>>>(
            tokp, decEmb_bf, decWih_bf, decWhh_bf, dec_bias, hp, hn, cbuf);
        float* outt = out + (size_t)t * BATCH * VOCAB;
        logits_kernel<<<500, 256, 0, stream>>>(hn, fcW_bf, fcb, outt);
        argmax_kernel<<<BATCH, 256, 0, stream>>>(outt, ntok);
        tokp = ntok;
        __bf16* tmp = hp; hp = hn; hn = tmp;
    }
}
